// CustomizableLRCLLoss_23776938950913
// MI455X (gfx1250) — matrix operand order repaired and checked by reading, not compiled, not run on hardware
//
#include <hip/hip_runtime.h>
#include <hip/hip_bf16.h>
#include <math.h>

// ---------------------------------------------------------------------------
// CustomizableLRCLLoss for MI455X (gfx1250, wave32)
//   B=32 rows, N=512, P=130816 pairs/row (tiles of 16 pairs per wave)
//   Coefficients live in the (constant) WMMA A operand; per-pair basis values
//   form the B operand, so D rows land per-pair in lanes -- no shuffle-back.
//   All transcendentals via hardware v_exp/v_log/v_rcp (branch-free).
// Workspace layout (floats):
//   [0      .. 16383]  s (normalized scores, B*N)
//   [16384  .. 17407]  per-(row,block) loss partial sums  (B * GX)
//   [17408  .. 18431]  per-(row,block) keep counts        (B * GX)
// ---------------------------------------------------------------------------

typedef __attribute__((ext_vector_type(16))) _Float16 v16h;
typedef __attribute__((ext_vector_type(8)))  float    v8f;

#define BATCH   32
#define NCOL    512
#define NPAIR   130816   // 512*511/2
#define NTILE   8176     // NPAIR / 16
#define GX      32       // blocks per row in pair kernel
#define TPB     256
#define FLOORW  0.001f
#define L2E     1.4426950408889634f
#define LN2     0.6931471805599453f

__device__ __forceinline__ float exp2hw(float x) { return __builtin_amdgcn_exp2f(x); }
__device__ __forceinline__ float log2hw(float x) { return __builtin_amdgcn_logf(x); }
__device__ __forceinline__ float rcphw(float x)  { return __builtin_amdgcn_rcpf(x); }

__device__ __forceinline__ float splus_fast(float z) {
    // softplus(z) = max(z,0) + ln2 * log2(1 + 2^(-|z|*log2e)), branch-free
    return fmaxf(z, 0.0f) + LN2 * log2hw(1.0f + exp2hw(-fabsf(z) * L2E));
}
__device__ __forceinline__ float maskf_of(float t) {
    // ~isclose(t, -1): |t+1| > atol + rtol*1  (atol=1e-8, rtol=1e-5)
    return (fabsf(t + 1.0f) > 1.00001e-5f) ? 1.0f : 0.0f;
}

// ---------------------------------------------------------------------------
// Kernel 1: per-row masked mean/var normalization of predictions -> s
// ---------------------------------------------------------------------------
__global__ __launch_bounds__(TPB) void lrcl_norm_kernel(
    const float* __restrict__ pred, const float* __restrict__ targ,
    float* __restrict__ srow)
{
    const int b = blockIdx.x;
    const int tid = threadIdx.x;
    __shared__ float r0[TPB];
    __shared__ float r1[TPB];

    const int i0 = tid, i1 = tid + TPB;
    const float t0 = targ[b * NCOL + i0];
    const float t1 = targ[b * NCOL + i1];
    const float m0 = maskf_of(t0);
    const float m1 = maskf_of(t1);
    const float x0 = m0 * pred[b * NCOL + i0];
    const float x1 = m1 * pred[b * NCOL + i1];

    r0[tid] = x0 + x1;      // masked sum
    r1[tid] = m0 + m1;      // mask count
    __syncthreads();
    for (int s = TPB / 2; s > 0; s >>= 1) {
        if (tid < s) { r0[tid] += r0[tid + s]; r1[tid] += r1[tid + s]; }
        __syncthreads();
    }
    const float denom = fmaxf(r1[0], 1.0f);
    const float mean  = r0[0] / denom;
    __syncthreads();

    const float c0 = x0 - mean;
    const float c1 = x1 - mean;
    r0[tid] = c0 * c0 * m0 + c1 * c1 * m1;
    __syncthreads();
    for (int s = TPB / 2; s > 0; s >>= 1) {
        if (tid < s) r0[tid] += r0[tid + s];
        __syncthreads();
    }
    const float rstd = rsqrtf(r0[0] / denom + 1e-6f);
    srow[b * NCOL + i0] = c0 * rstd;
    srow[b * NCOL + i1] = c1 * rstd;
}

// ---------------------------------------------------------------------------
// Kernel 2: pair-tile loss. grid = (GX, B). Each wave owns tiles of 16 pairs.
//
//   WMMA #1: D1 = A1 x B1
//     A1 (const, 16x32): row0 = c_tau @ K0..7, row1 = c_w @ K8..13
//     B1 (per tile, 32x16): col p = pair p basis:
//        lane p    elems 0..7 = softplus tau basis (K0..7)
//        lane p+16 elems 0..5 = sigmoid  w   basis (K8..13)
//     -> d1[0] = tau(pair lane), d1[1] = w_dot(pair lane)  [lanes 0..15]
//
//   WMMA #2: D2 = A2 x B2
//     A2 (const): row0 = c_g[0..3] @ K0..3, c_g[4..7] @ K8..11
//     B2 (per tile): lane p elems 0..3 = g basis k0..3 (K0..3),
//                    lane p+16 elems 0..3 = g basis k4..7 (K8..11),
//                    both scaled by keepf
//     -> d2[0] = g(pair lane)   [lanes 0..15]
//
//   pair_loss = (d1[1] + FLOOR) * d2[0], owned by lanes 0..15.
// ---------------------------------------------------------------------------
__global__ __launch_bounds__(TPB) void lrcl_pair_kernel(
    const float* __restrict__ targ, const float* __restrict__ srow,
    const float* __restrict__ th_tau, const float* __restrict__ th_g,
    const float* __restrict__ th_w,
    float* __restrict__ psum, float* __restrict__ pcnt)
{
    const int b    = blockIdx.y;
    const int tid  = threadIdx.x;
    const int lane = tid & 31;
    const bool low = lane < 16;
    const int wavesPerRow = gridDim.x * (TPB >> 5);
    const int waveId = blockIdx.x * (TPB >> 5) + (tid >> 5);

    // learned nonnegative coefficients (uniform per thread)
    float ctau[8], cg[8], cw[6];
#pragma unroll
    for (int k = 0; k < 8; ++k) ctau[k] = splus_fast(th_tau[k]);
#pragma unroll
    for (int k = 0; k < 8; ++k) cg[k] = splus_fast(th_g[k]);
#pragma unroll
    for (int k = 0; k < 6; ++k) cw[k] = splus_fast(th_w[k]);

    // Constant A matrices, ISA 16-bit A layout:
    //   lanes 0-15: row M=lane, elems 0..7 = K0..7, elems 8..15 = K16..23
    //   lanes 16-31: row M=lane-16, elems 0..7 = K8..15, elems 8..15 = K24..31
    v16h am1 = {};
    v16h am2 = {};
    if (lane == 0) {                       // row 0: c_tau @ K0..7
#pragma unroll
        for (int k = 0; k < 8; ++k) am1[k] = (_Float16)ctau[k];
#pragma unroll
        for (int k = 0; k < 4; ++k) {      // row 0: c_g split K0..3 / K8..11
            am2[k]     = (_Float16)cg[k];
        }
    }
    if (lane == 16) {
#pragma unroll
        for (int k = 0; k < 4; ++k) am2[k] = (_Float16)cg[4 + k];
    }
    if (lane == 17) {                      // row 1: c_w @ K8..13
#pragma unroll
        for (int k = 0; k < 6; ++k) am1[k] = (_Float16)cw[k];
    }

    const float* tg = targ + b * NCOL;
    const float* sc = srow + b * NCOL;
    __builtin_prefetch(tg, 0, 0);          // global_prefetch_b8
    __builtin_prefetch(sc, 0, 0);

    const int   pl   = lane & 15;          // pair slot this lane serves
    const float ownf = low ? 1.0f : 0.0f;  // lanes 0..15 own the pair results

    float acc = 0.0f, cnt = 0.0f;
    const v8f zacc = {};

    for (int tile = waveId; tile < NTILE; tile += wavesPerRow) {
        const int p = tile * 16 + pl;

        // invert triangular pair index: i < j, row-major upper triangle
        int i = (int)((1023.0f - __builtin_amdgcn_sqrtf(1046529.0f - 8.0f * (float)p)) * 0.5f);
        i = (i < 0) ? 0 : ((i > NCOL - 2) ? NCOL - 2 : i);
        int base = (i * (2 * NCOL - 1 - i)) >> 1;
        while (base > p) { --i; base = (i * (2 * NCOL - 1 - i)) >> 1; }
        while (i < NCOL - 2) {
            const int nb = ((i + 1) * (2 * NCOL - 2 - i)) >> 1;
            if (nb <= p) { ++i; base = nb; } else break;
        }
        const int j = i + 1 + (p - base);

        const float ri = tg[i], rj = tg[j];
        const float dr = ri - rj;
        const float tsgn = (dr > 0.0f) ? 1.0f : ((dr < 0.0f) ? -1.0f : 0.0f);
        const float keepf = ((tsgn != 0.0f) &&
                             (maskf_of(ri) > 0.5f) &&
                             (maskf_of(rj) > 0.5f)) ? 1.0f : 0.0f;
        const float dsv = sc[i] - sc[j];
        const float adr = fabsf(dr);

        // B1 fill: branch-free, per-lane select between softplus(tau grid)
        // and sigmoid(w grid); shared hardware exp2 between both formulas.
        // B layout: lane p = col p holding K0..7 in elems 0..7, K16..23 in 8..15,
        //           lane p+16 = col p holding K8..15, K24..31.
        v16h b1 = {};
#pragma unroll
        for (int k = 0; k < 8; ++k) {
            const float ssp = 0.5f + (float)k * (3.5f / 7.0f);
            const float bsp = -2.0f + (float)k * (4.0f / 7.0f);
            const float ssg = (k < 6) ? (0.5f + (float)k * (3.5f / 5.0f)) : 0.0f;
            const float bsg = (k < 6) ? (-2.0f + (float)k * (4.0f / 5.0f)) : 0.0f;
            const float z  = adr * (low ? ssp : ssg) + (low ? bsp : bsg);
            const float e  = exp2hw((low ? -fabsf(z) : -z) * L2E);
            const float sp = fmaxf(z, 0.0f) + LN2 * log2hw(1.0f + e);
            const float sg = rcphw(1.0f + e);
            const float val = low ? sp : ((k < 6) ? sg : 0.0f);
            b1[k] = (_Float16)val;
        }
        v8f d1 = __builtin_amdgcn_wmma_f32_16x16x32_f16(
            false, am1, false, b1, (short)0, zacc, false, false);

        // tau for pair pl: d1[0] at lane pl; one broadcast feeds both halves
        const float tauv = __shfl(d1[0], pl, 32);
        const float m = tauv - tsgn * dsv;

        // B2: g basis of margin, split 4+4 across the two lane halves
        v16h b2 = {};
#pragma unroll
        for (int k = 0; k < 4; ++k) {
            const float sl = low ? (0.5f + (float)k * (3.5f / 7.0f))
                                 : (0.5f + (float)(k + 4) * (3.5f / 7.0f));
            const float bi = low ? (-2.0f + (float)k * (4.0f / 7.0f))
                                 : (-2.0f + (float)(k + 4) * (4.0f / 7.0f));
            b2[k] = (_Float16)(keepf * splus_fast(m * sl + bi));
        }
        v8f d2 = __builtin_amdgcn_wmma_f32_16x16x32_f16(
            false, am2, false, b2, (short)0, zacc, false, false);

        // pair_loss for pair pl sits at lane pl: w in d1[1], g in d2[0]
        acc += ownf * ((d1[1] + FLOORW) * d2[0]);
        cnt += ownf * keepf;
    }

    // deterministic block tree reduction
    __shared__ float ssum[TPB];
    __shared__ float scnt[TPB];
    ssum[tid] = acc;
    scnt[tid] = cnt;
    __syncthreads();
    for (int s = TPB / 2; s > 0; s >>= 1) {
        if (tid < s) { ssum[tid] += ssum[tid + s]; scnt[tid] += scnt[tid + s]; }
        __syncthreads();
    }
    if (tid == 0) {
        psum[b * gridDim.x + blockIdx.x] = ssum[0];
        pcnt[b * gridDim.x + blockIdx.x] = scnt[0];
    }
}

// ---------------------------------------------------------------------------
// Kernel 3: fixed-order final reduction, single wave. lane b owns row b.
// ---------------------------------------------------------------------------
__global__ __launch_bounds__(32) void lrcl_final_kernel(
    const float* __restrict__ psum, const float* __restrict__ pcnt,
    float* __restrict__ out)
{
    const int b = threadIdx.x;   // 0..31
    float rs = 0.0f, rc = 0.0f;
    for (int k = 0; k < GX; ++k) {
        rs += psum[b * GX + k];
        rc += pcnt[b * GX + k];
    }
    const float rowLoss = rs / fmaxf(rc, 1.0f);
    const float valid   = (rc > 0.0f) ? 1.0f : 0.0f;
    float num = rowLoss * valid;
    float den = valid;
#pragma unroll
    for (int o = 16; o > 0; o >>= 1) {
        num += __shfl_down(num, o, 32);
        den += __shfl_down(den, o, 32);
    }
    if (b == 0) out[0] = num / fmaxf(den, 1.0f);
}

// ---------------------------------------------------------------------------
extern "C" void kernel_launch(void* const* d_in, const int* in_sizes, int n_in,
                              void* d_out, int out_size, void* d_ws, size_t ws_size,
                              hipStream_t stream) {
    (void)in_sizes; (void)n_in; (void)out_size; (void)ws_size;
    const float* pred   = (const float*)d_in[0];
    const float* targ   = (const float*)d_in[1];
    const float* th_tau = (const float*)d_in[2];
    const float* th_g   = (const float*)d_in[3];
    const float* th_w   = (const float*)d_in[4];
    float* out = (float*)d_out;

    float* ws   = (float*)d_ws;
    float* srow = ws;                          // B*N
    float* psum = ws + BATCH * NCOL;           // B*GX
    float* pcnt = psum + BATCH * GX;           // B*GX

    lrcl_norm_kernel<<<dim3(BATCH), dim3(TPB), 0, stream>>>(pred, targ, srow);
    lrcl_pair_kernel<<<dim3(GX, BATCH), dim3(TPB), 0, stream>>>(
        targ, srow, th_tau, th_g, th_w, psum, pcnt);
    lrcl_final_kernel<<<dim3(1), dim3(32), 0, stream>>>(psum, pcnt, out);
}
